// WindowAttention_20942260535752
// MI455X (gfx1250) — compile-verified, matrix operand order checked by reading, not run on hardware
//
#include <hip/hip_runtime.h>
#include <hip/hip_bf16.h>

typedef __attribute__((ext_vector_type(16))) _Float16 v16h;
typedef __attribute__((ext_vector_type(8)))  _Float16 v8h;
typedef __attribute__((ext_vector_type(8)))  float    v8f;

// Load a 16x32 f16 WMMA fragment (A or B^T) from a row-major matrix.
// Per CDNA5 ISA 7.12.2 (16-bit A-matrix 16x32): lane = 16*half + m holds
// elements 0..7  -> K = col0 + 8*half + {0..7}
// elements 8..15 -> K = col0 + 16 + 8*half + {0..7}
__device__ __forceinline__ v16h load_frag(const _Float16* __restrict__ base,
                                          int row0, int ld, int col0, int lane) {
    const int m  = lane & 15;
    const int hf = lane >> 4;
    const _Float16* p = base + (row0 + m) * ld + col0 + hf * 8;
    v16h f;
    ((v8h*)&f)[0] = *(const v8h*)(p);
    ((v8h*)&f)[1] = *(const v8h*)(p + 16);
    return f;
}

__device__ __forceinline__ v8f wmma_step(v16h a, v16h b, v8f c) {
    return __builtin_amdgcn_wmma_f32_16x16x32_f16(
        /*neg_a=*/false, a, /*neg_b=*/false, b,
        /*c_mod=*/(short)0, c, /*reuse_a=*/false, /*reuse_b=*/false);
}

// ---------------------------------------------------------------------------
// Prep: convert Wq (65536), Wkv (131072), Wp (65536) f32 -> f16 into d_ws.
// Layout in ws (halfs): [0,65536)=WqH, [65536,196608)=WkvH, [196608,262144)=WpH
// ---------------------------------------------------------------------------
__global__ void cvt_weights_kernel(const float* __restrict__ wq,
                                   const float* __restrict__ wkv,
                                   const float* __restrict__ wp,
                                   _Float16* __restrict__ ws) {
    int i = blockIdx.x * 256 + threadIdx.x;   // 0 .. 262143
    float v;
    if (i < 65536)       v = wq[i];
    else if (i < 196608) v = wkv[i - 65536];
    else                 v = wp[i - 196608];
    ws[i] = (_Float16)v;
}

// ---------------------------------------------------------------------------
// Fused window attention: one workgroup (8 wave32) per window of 64 tokens.
// ---------------------------------------------------------------------------
__global__ __launch_bounds__(256) void win_attn_kernel(
    const float* __restrict__ x,      // [4096,64,256]
    const float* __restrict__ mask,   // [1024,64,64]
    const float* __restrict__ bq,     // [256]
    const float* __restrict__ bkv,    // [512]
    const float* __restrict__ rpb,    // [225,8]
    const float* __restrict__ bp,     // [256]
    const _Float16* __restrict__ wqh, // [256,256] f16
    const _Float16* __restrict__ wkvh,// [512,256] f16
    const _Float16* __restrict__ wph, // [256,256] f16
    float* __restrict__ out)          // [4096,64,256]
{
    __shared__ _Float16 xh[64 * 256];   // x in f16; reused as attn-out buffer
    __shared__ _Float16 qh[64 * 256];   // q * scale (f16)
    __shared__ _Float16 kh[64 * 256];   // k (f16)
    __shared__ _Float16 vT[256 * 64];   // v transposed: [channel][token]
    __shared__ float    sS[64 * 64];    // per-head scores (f32)
    __shared__ _Float16 sP[64 * 64];    // per-head probs (f16)
    __shared__ float    rmax[64][4];
    __shared__ float    rsum[64][4];

    const int tid  = threadIdx.x;
    const int lane = tid & 31;
    const int wave = tid >> 5;
    const int m    = lane & 15;
    const int hf   = lane >> 4;
    const int b    = blockIdx.x;
    const float scale = 0.17677669529663687f;   // 1/sqrt(32)

    // ---- Phase 1: load x[b] (64x256 f32) -> f16 LDS -------------------------
    {
        const float4* xp = (const float4*)(x + (size_t)b * 16384);
        for (int i = tid; i < 4096; i += 256) {
            float4 f = xp[i];
            _Float16* d = xh + i * 4;
            d[0] = (_Float16)f.x; d[1] = (_Float16)f.y;
            d[2] = (_Float16)f.z; d[3] = (_Float16)f.w;
        }
    }
    __syncthreads();

    // ---- Phase 2: q/k/v projections (192 16x16 tiles, K=256) ----------------
    for (int t = wave; t < 192; t += 8) {
        const int mat = t >> 6;            // 0=q, 1=k, 2=v
        const int rem = t & 63;
        const int r = (rem >> 4) * 16;     // token-row base
        const int c = (rem & 15) * 16;     // out-channel base
        const _Float16* wbase =
            (mat == 0) ? wqh : ((mat == 1) ? wkvh : (wkvh + 65536));
        v8f acc = {};
        for (int kk = 0; kk < 256; kk += 32) {
            v16h a = load_frag(xh,    r, 256, kk, lane);
            v16h w = load_frag(wbase, c, 256, kk, lane);
            acc = wmma_step(a, w, acc);
        }
        const int cg = c + m;              // this lane's output channel
        if (mat == 0) {
            float bb = bq[cg];
            for (int i = 0; i < 8; ++i)
                qh[(r + hf * 8 + i) * 256 + cg] = (_Float16)((acc[i] + bb) * scale);
        } else if (mat == 1) {
            float bb = bkv[cg];
            for (int i = 0; i < 8; ++i)
                kh[(r + hf * 8 + i) * 256 + cg] = (_Float16)(acc[i] + bb);
        } else {
            float bb = bkv[256 + cg];
            for (int i = 0; i < 8; ++i)    // store transposed: vT[chan][token]
                vT[cg * 64 + r + hf * 8 + i] = (_Float16)(acc[i] + bb);
        }
    }
    __syncthreads();

    // softmax work split: 4 threads per row
    const int srow = tid >> 2;             // 0..63
    const int sj0  = (tid & 3) * 16;
    const int sseg = tid & 3;
    const int mbase = (b & 1023) * 4096 + srow * 64;
    const int ih = srow >> 3, iw = srow & 7;

    // ---- Phase 3: per-head attention ---------------------------------------
    for (int h = 0; h < 8; ++h) {
        // S = (q*scale) @ k^T : 16 tiles, 2 per wave, one WMMA each (K=32=d)
        for (int q2 = 0; q2 < 2; ++q2) {
            const int t = wave * 2 + q2;
            const int r = (t >> 2) * 16, c = (t & 3) * 16;
            v16h a  = load_frag(qh, r, 256, h * 32, lane);
            v16h kb = load_frag(kh, c, 256, h * 32, lane);
            v8f acc = {};
            acc = wmma_step(a, kb, acc);
            for (int i = 0; i < 8; ++i)
                sS[(r + hf * 8 + i) * 64 + c + m] = acc[i];
        }
        __syncthreads();

        // softmax A: add rel-pos bias + shift mask, partial row max
        {
            float pm = -3.0e38f;
            for (int j = sj0; j < sj0 + 16; ++j) {
                const int jh = j >> 3, jw = j & 7;
                const int idx = (ih - jh + 7) * 15 + (iw - jw + 7);
                float v = sS[srow * 64 + j] + rpb[idx * 8 + h] + mask[mbase + j];
                sS[srow * 64 + j] = v;
                pm = fmaxf(pm, v);
            }
            rmax[srow][sseg] = pm;
        }
        __syncthreads();
        // softmax B: exp(x - rowmax), partial sum
        {
            float rm = fmaxf(fmaxf(rmax[srow][0], rmax[srow][1]),
                             fmaxf(rmax[srow][2], rmax[srow][3]));
            float ps = 0.f;
            for (int j = sj0; j < sj0 + 16; ++j) {
                float e = __expf(sS[srow * 64 + j] - rm);
                sS[srow * 64 + j] = e;
                ps += e;
            }
            rsum[srow][sseg] = ps;
        }
        __syncthreads();
        // softmax C: normalize -> f16 probs
        {
            float rs = rsum[srow][0] + rsum[srow][1] + rsum[srow][2] + rsum[srow][3];
            float inv = 1.f / rs;
            for (int j = sj0; j < sj0 + 16; ++j)
                sP[srow * 64 + j] = (_Float16)(sS[srow * 64 + j] * inv);
        }
        __syncthreads();

        // out_h = P @ v_h : 8 tiles, 1 per wave, K=64 -> 2 WMMA
        {
            const int r = (wave >> 1) * 16;
            const int c = (wave & 1) * 16;
            v8f acc = {};
            for (int kk = 0; kk < 64; kk += 32) {
                v16h a  = load_frag(sP, r, 64, kk, lane);
                v16h vv = load_frag(vT, h * 32 + c, 64, kk, lane);
                acc = wmma_step(a, vv, acc);
            }
            for (int i = 0; i < 8; ++i)   // reuse xh as attention-output buffer
                xh[(r + hf * 8 + i) * 256 + h * 32 + c + m] = (_Float16)acc[i];
        }
        __syncthreads();
    }

    // ---- Phase 4: output projection y = attn_out @ Wp^T + bp ----------------
    float* outb = out + (size_t)b * 16384;
    for (int t = wave; t < 64; t += 8) {
        const int r = (t >> 4) * 16;
        const int c = (t & 15) * 16;
        v8f acc = {};
        for (int kk = 0; kk < 256; kk += 32) {
            v16h a = load_frag(xh,  r, 256, kk, lane);
            v16h w = load_frag(wph, c, 256, kk, lane);
            acc = wmma_step(a, w, acc);
        }
        const int cg = c + m;
        float bb = bp[cg];
        for (int i = 0; i < 8; ++i)
            outb[(size_t)(r + hf * 8 + i) * 256 + cg] = acc[i] + bb;
    }
}

extern "C" void kernel_launch(void* const* d_in, const int* in_sizes, int n_in,
                              void* d_out, int out_size, void* d_ws, size_t ws_size,
                              hipStream_t stream) {
    const float* x    = (const float*)d_in[0];
    const float* mask = (const float*)d_in[1];
    const float* Wq   = (const float*)d_in[2];
    const float* bq   = (const float*)d_in[3];
    const float* Wkv  = (const float*)d_in[4];
    const float* bkv  = (const float*)d_in[5];
    const float* rpb  = (const float*)d_in[6];
    const float* Wp   = (const float*)d_in[7];
    const float* bp   = (const float*)d_in[8];
    float* out = (float*)d_out;
    _Float16* ws = (_Float16*)d_ws;   // needs 262144 halfs = 512 KB

    cvt_weights_kernel<<<1024, 256, 0, stream>>>(Wq, Wkv, Wp, ws);
    win_attn_kernel<<<4096, 256, 0, stream>>>(
        x, mask, bq, bkv, rpb, bp,
        ws, ws + 65536, ws + 196608, out);
}
